// GraphMDANet_52020643889238
// MI455X (gfx1250) — compile-verified, matrix operand order checked by reading, not run on hardware
//
#include <hip/hip_runtime.h>
#include <hip/hip_bf16.h>

// ---------------------------------------------------------------------------
// GraphMDANet forward on gfx1250 (MI455X, wave32, WMMA + async LDS copies).
//   1) f32->f16 conversions; weights transposed to [N,K] so BOTH WMMA operand
//      fragments load as contiguous 16B global_load_b128
//   2) h1 = relu(x@W0+b0)        WMMA 32x32-tile GEMM (f16 out)
//   3) h2 = relu(h1@W1+b1)       WMMA GEMM (f32 + f16 out)
//   4) sq[i] = ||h2_i||^2
//   5) d2 = sq_i + sq_j - 2*h2@h2^T (+1e9 diag)  WMMA + prefetch
//   6) per-row top-K=32 in LDS; row staged via global_load_async_to_lds_b128
//   7) Wh = h2@gat_W             WMMA GEMM
//   8) f1/f2 attention projections
//   9) sparse masked softmax over 33 neighbors + elu aggregate -> hp
//  10) head: pre-reduced (over 8 tiled heads) cls/dsc weights, 22 outs/row
// ---------------------------------------------------------------------------

#define N_NODES 6144
#define D_IN    512
#define H1_DIM  256
#define H2_DIM  128
#define G_DIM   64
#define K_NN    32
#define N_CLS   16
#define N_DOM   3
#define OUT_W   (N_CLS + N_DOM * 2)   // 22
#define LEAKY_ALPHA 0.2f

typedef __attribute__((ext_vector_type(16))) _Float16 v16h;
typedef __attribute__((ext_vector_type(8)))  float    v8f;

// ---------------- WMMA fragment helper (ISA 7.12.2 layout) ------------------
__device__ inline v16h load_frag(const _Float16* __restrict__ A, int ld,
                                 int row, int k0, int lane) {
    const int kA = k0 + ((lane & 16) ? 8 : 0);
    const _Float16* p = A + (size_t)row * ld + kA;
    v16h a;
#pragma unroll
    for (int t = 0; t < 8; ++t) {
        a[t]     = p[t];
        a[8 + t] = p[16 + t];
    }
    return a;
}

__device__ inline v8f wmma_f16(v16h a, v16h b, v8f c) {
    return __builtin_amdgcn_wmma_f32_16x16x32_f16(
        /*neg_a=*/false, a, /*neg_b=*/false, b,
        /*c_mod=*/(short)0, c, /*reuse_a=*/false, /*reuse_b=*/false);
}

// ---------------------------- conversion kernels ----------------------------
__global__ void cvt_f32_to_f16(const float* __restrict__ in,
                               _Float16* __restrict__ out, int n) {
    int i = blockIdx.x * blockDim.x + threadIdx.x;
    if (i < n) out[i] = (_Float16)in[i];
}

// in: [K,N] row-major f32  ->  out: [N,K] row-major f16 (transposed)
__global__ void cvt_transpose_f16(const float* __restrict__ in,
                                  _Float16* __restrict__ out, int K, int N) {
    int i = blockIdx.x * blockDim.x + threadIdx.x;
    if (i >= K * N) return;
    int k = i / N, n = i % N;
    out[(size_t)n * K + k] = (_Float16)in[i];
}

// ------------------------- 32x32-tile WMMA GEMM -----------------------------
__device__ inline void store_gemm_tile(v8f c, int mbase, int nbase,
                                       const float* __restrict__ bias,
                                       float* __restrict__ Cf,
                                       _Float16* __restrict__ Ch,
                                       int N, int relu, int lane) {
    const int col = nbase + (lane & 15);
    const int rb  = mbase + ((lane & 16) ? 8 : 0);
    const float bv = bias ? bias[col] : 0.0f;
#pragma unroll
    for (int r = 0; r < 8; ++r) {
        const int row = rb + r;
        float v = c[r] + bv;
        if (relu) v = v > 0.0f ? v : 0.0f;
        if (Cf) Cf[(size_t)row * N + col] = v;
        if (Ch) Ch[(size_t)row * N + col] = (_Float16)v;
    }
}

__global__ __launch_bounds__(32)
void wmma_gemm32(const _Float16* __restrict__ A, const _Float16* __restrict__ Bt,
                 float* __restrict__ Cf, _Float16* __restrict__ Ch,
                 const float* __restrict__ bias,
                 int M, int N, int K, int relu) {
    const int lane = threadIdx.x;
    const int m0 = blockIdx.x * 32;
    const int n0 = blockIdx.y * 32;

    v8f c00 = {}, c01 = {}, c10 = {}, c11 = {};
    for (int k0 = 0; k0 < K; k0 += 32) {
        v16h a0 = load_frag(A,  K, m0 + (lane & 15),      k0, lane);
        v16h a1 = load_frag(A,  K, m0 + 16 + (lane & 15), k0, lane);
        v16h b0 = load_frag(Bt, K, n0 + (lane & 15),      k0, lane);
        v16h b1 = load_frag(Bt, K, n0 + 16 + (lane & 15), k0, lane);
        c00 = wmma_f16(a0, b0, c00);
        c01 = wmma_f16(a0, b1, c01);
        c10 = wmma_f16(a1, b0, c10);
        c11 = wmma_f16(a1, b1, c11);
    }
    store_gemm_tile(c00, m0,      n0,      bias, Cf, Ch, N, relu, lane);
    store_gemm_tile(c01, m0,      n0 + 16, bias, Cf, Ch, N, relu, lane);
    store_gemm_tile(c10, m0 + 16, n0,      bias, Cf, Ch, N, relu, lane);
    store_gemm_tile(c11, m0 + 16, n0 + 16, bias, Cf, Ch, N, relu, lane);
}

// --------------------------- row squared norms ------------------------------
__global__ void row_sqnorm(const float* __restrict__ h, float* __restrict__ sq,
                           int N, int D) {
    int i = blockIdx.x * blockDim.x + threadIdx.x;
    if (i >= N) return;
    float s = 0.0f;
    const float* p = h + (size_t)i * D;
    for (int d = 0; d < D; ++d) s += p[d] * p[d];
    sq[i] = s;
}

// --------------- distance matrix: d2 = sq_i + sq_j - 2 h h^T ----------------
__device__ inline void store_dist_tile(v8f c, int mbase, int nbase,
                                       const float* __restrict__ sq,
                                       float* __restrict__ d2, int N, int lane) {
    const int col = nbase + (lane & 15);
    const int rb  = mbase + ((lane & 16) ? 8 : 0);
    const float sc = sq[col];
#pragma unroll
    for (int r = 0; r < 8; ++r) {
        const int row = rb + r;
        float v = sq[row] + sc - 2.0f * c[r];
        if (row == col) v += 1e9f;   // exclude self from kNN
        d2[(size_t)row * N + col] = v;
    }
}

__global__ __launch_bounds__(32)
void wmma_dist(const _Float16* __restrict__ H, const float* __restrict__ sq,
               float* __restrict__ d2, int N, int K) {
    const int lane = threadIdx.x;
    const int m0 = blockIdx.x * 32;
    const int n0 = blockIdx.y * 32;

    v8f c00 = {}, c01 = {}, c10 = {}, c11 = {};
    for (int k0 = 0; k0 < K; k0 += 32) {
        if (k0 + 32 < K) {  // L2 prefetch of next k-slab (global_prefetch_b8)
            __builtin_prefetch(H + (size_t)(m0 + (lane & 15)) * K + k0 + 32, 0, 0);
            __builtin_prefetch(H + (size_t)(n0 + (lane & 15)) * K + k0 + 32, 0, 0);
        }
        v16h a0 = load_frag(H, K, m0 + (lane & 15),      k0, lane);
        v16h a1 = load_frag(H, K, m0 + 16 + (lane & 15), k0, lane);
        v16h b0 = load_frag(H, K, n0 + (lane & 15),      k0, lane);
        v16h b1 = load_frag(H, K, n0 + 16 + (lane & 15), k0, lane);
        c00 = wmma_f16(a0, b0, c00);
        c01 = wmma_f16(a0, b1, c01);
        c10 = wmma_f16(a1, b0, c10);
        c11 = wmma_f16(a1, b1, c11);
    }
    store_dist_tile(c00, m0,      n0,      sq, d2, N, lane);
    store_dist_tile(c01, m0,      n0 + 16, sq, d2, N, lane);
    store_dist_tile(c10, m0 + 16, n0,      sq, d2, N, lane);
    store_dist_tile(c11, m0 + 16, n0 + 16, sq, d2, N, lane);
}

// ------------------------------ per-row top-K -------------------------------
// Stage the 24KB row into LDS with the CDNA5 async copy path (ASYNCcnt),
// bypassing VGPRs, then run 32 argmin selection passes.
__global__ __launch_bounds__(256)
void topk_rows(const float* __restrict__ d2, int* __restrict__ idx,
               int N, int Ksel) {
    __shared__ float srow[N_NODES];
    __shared__ float rmin[256];
    __shared__ int   rid[256];
    const int row = blockIdx.x;
    const int t   = threadIdx.x;

    // ---- async global -> LDS staging (16B per lane per issue) ----
    for (int j4 = t; j4 < N / 4; j4 += 256) {
        // Hardware LDS address = low 32 bits of the flat shared address
        // (ISA 10.2 aperture rule).
        unsigned int lds_off = (unsigned int)(unsigned long long)(&srow[j4 * 4]);
        const float* gp = d2 + (size_t)row * N + (size_t)j4 * 4;
        asm volatile("global_load_async_to_lds_b128 %0, %1, off"
                     :: "v"(lds_off), "v"(gp)
                     : "memory");
    }
    asm volatile("s_wait_asynccnt 0x0" ::: "memory");
    __syncthreads();

    for (int k = 0; k < Ksel; ++k) {
        float best = 3.0e38f;
        int   bi   = 0x7fffffff;
        for (int j = t; j < N; j += 256) {
            float v = srow[j];
            if (v < best || (v == best && j < bi)) { best = v; bi = j; }
        }
        rmin[t] = best; rid[t] = bi;
        __syncthreads();
        for (int s = 128; s > 0; s >>= 1) {
            if (t < s) {
                if (rmin[t + s] < rmin[t] ||
                    (rmin[t + s] == rmin[t] && rid[t + s] < rid[t])) {
                    rmin[t] = rmin[t + s]; rid[t] = rid[t + s];
                }
            }
            __syncthreads();
        }
        if (t == 0) {
            idx[row * Ksel + k] = rid[0];
            srow[rid[0]] = 3.0e38f;   // remove selected
        }
        __syncthreads();
    }
}

// ------------------- attention projections f1, f2 ---------------------------
__global__ void attn_proj(const float* __restrict__ Wh,
                          const float* __restrict__ a,     // [2G]
                          float* __restrict__ f1, float* __restrict__ f2,
                          int N, int G) {
    int i = blockIdx.x * blockDim.x + threadIdx.x;
    if (i >= N) return;
    const float* p = Wh + (size_t)i * G;
    float s1 = 0.0f, s2 = 0.0f;
    for (int g = 0; g < G; ++g) {
        s1 += p[g] * a[g];
        s2 += p[g] * a[G + g];
    }
    f1[i] = s1; f2[i] = s2;
}

// ---- sparse masked softmax over {knn(i), i} + elu(att @ Wh) aggregate ------
__global__ __launch_bounds__(64)
void gat_aggregate(const float* __restrict__ Wh, const float* __restrict__ f1,
                   const float* __restrict__ f2, const int* __restrict__ idx,
                   float* __restrict__ hp, int N, int G, int Ksel) {
    __shared__ float att[K_NN + 1];
    __shared__ int   nbr[K_NN + 1];
    const int row = blockIdx.x;
    const int t   = threadIdx.x;
    const int M   = Ksel + 1;   // 33 members incl. self loop

    if (t < M) {
        int j = (t < Ksel) ? idx[row * Ksel + t] : row;
        nbr[t] = j;
        float e = f1[row] + f2[j];
        att[t] = e > 0.0f ? e : LEAKY_ALPHA * e;   // leaky_relu
    }
    __syncthreads();
    if (t == 0) {
        float mx = -3.0e38f;
        for (int j = 0; j < M; ++j) mx = fmaxf(mx, att[j]);
        float s = 0.0f;
        for (int j = 0; j < M; ++j) { float v = __expf(att[j] - mx); att[j] = v; s += v; }
        float inv = 1.0f / s;
        for (int j = 0; j < M; ++j) att[j] *= inv;
    }
    __syncthreads();
    if (t < G) {
        float acc = 0.0f;
        for (int j = 0; j < M; ++j)
            acc += att[j] * Wh[(size_t)nbr[j] * G + t];
        hp[(size_t)row * G + t] = acc > 0.0f ? acc : (__expf(acc) - 1.0f); // elu
    }
}

// ---- pre-reduce head weights over the 8 tiled (shared) attention heads -----
__global__ void reduce_heads(const float* __restrict__ clsW,   // [8*64,16]
                             const float* __restrict__ dscW,   // [3,8*64,2]
                             float* __restrict__ WcS,          // [64,16]
                             float* __restrict__ WdS) {        // [3,64,2]
    int t = blockIdx.x * blockDim.x + threadIdx.x;
    if (t < G_DIM * N_CLS) {
        int g = t / N_CLS, c = t % N_CLS;
        float s = 0.0f;
        for (int h = 0; h < 8; ++h) s += clsW[(h * G_DIM + g) * N_CLS + c];
        WcS[t] = s;
    }
    if (t < N_DOM * G_DIM * 2) {
        int k = t / (G_DIM * 2);
        int r = t % (G_DIM * 2);
        int g = r / 2, o = r % 2;
        float s = 0.0f;
        for (int h = 0; h < 8; ++h)
            s += dscW[((size_t)k * (8 * G_DIM) + h * G_DIM + g) * 2 + o];
        WdS[t] = s;
    }
}

// ------------------ fused class + discriminator head ------------------------
__global__ __launch_bounds__(32)
void head_kernel(const float* __restrict__ hp, const float* __restrict__ WcS,
                 const float* __restrict__ WdS, const float* __restrict__ clsb,
                 const float* __restrict__ dscb, float* __restrict__ out,
                 int N, int G) {
    const int row = blockIdx.x;
    const int t   = threadIdx.x;
    const float* p = hp + (size_t)row * G;
    if (t < N_CLS) {
        float s = clsb[t];
        for (int g = 0; g < G; ++g) s += p[g] * WcS[g * N_CLS + t];
        out[(size_t)row * OUT_W + t] = s;
    } else if (t < OUT_W) {
        int q = t - N_CLS;
        int k = q / 2, o = q % 2;
        float s = dscb[k * 2 + o];
        for (int g = 0; g < G; ++g) s += p[g] * WdS[k * (G * 2) + g * 2 + o];
        out[(size_t)row * OUT_W + t] = s;
    }
}

// ---------------------------------------------------------------------------
static inline void* carve(char*& cur, size_t bytes) {
    void* p = (void*)cur;
    cur += (bytes + 255) & ~(size_t)255;
    return p;
}

extern "C" void kernel_launch(void* const* d_in, const int* in_sizes, int n_in,
                              void* d_out, int out_size, void* d_ws, size_t ws_size,
                              hipStream_t stream) {
    (void)in_sizes; (void)n_in; (void)out_size; (void)ws_size;
    const float* x     = (const float*)d_in[0];   // [6144,512]
    const float* fcW0  = (const float*)d_in[1];   // [512,256]
    const float* fcb0  = (const float*)d_in[2];   // [256]
    const float* fcW1  = (const float*)d_in[3];   // [256,128]
    const float* fcb1  = (const float*)d_in[4];   // [128]
    const float* gatW  = (const float*)d_in[5];   // [128,64]
    const float* gata  = (const float*)d_in[6];   // [128,1]
    const float* clsW  = (const float*)d_in[7];   // [512,16]
    const float* clsb  = (const float*)d_in[8];   // [16]
    const float* dscW  = (const float*)d_in[9];   // [3,512,2]
    const float* dscb  = (const float*)d_in[10];  // [3,2]
    float* out = (float*)d_out;                   // [6144,22]

    const int N = N_NODES;

    // ---- workspace carve-out (transposed f16 weights: [N,K] layout) ---------
    char* cur = (char*)d_ws;
    _Float16* x_h    = (_Float16*)carve(cur, (size_t)N * D_IN * 2);
    _Float16* W0t_h  = (_Float16*)carve(cur, (size_t)D_IN * H1_DIM * 2);   // [256,512]
    _Float16* h1_h   = (_Float16*)carve(cur, (size_t)N * H1_DIM * 2);
    _Float16* W1t_h  = (_Float16*)carve(cur, (size_t)H1_DIM * H2_DIM * 2); // [128,256]
    float*    h2_f   = (float*)   carve(cur, (size_t)N * H2_DIM * 4);
    _Float16* h2_h   = (_Float16*)carve(cur, (size_t)N * H2_DIM * 2);
    _Float16* gWt_h  = (_Float16*)carve(cur, (size_t)H2_DIM * G_DIM * 2);  // [64,128]
    float*    Wh_f   = (float*)   carve(cur, (size_t)N * G_DIM * 4);
    float*    sq     = (float*)   carve(cur, (size_t)N * 4);
    float*    f1     = (float*)   carve(cur, (size_t)N * 4);
    float*    f2     = (float*)   carve(cur, (size_t)N * 4);
    int*      knn    = (int*)     carve(cur, (size_t)N * K_NN * 4);
    float*    hp     = (float*)   carve(cur, (size_t)N * G_DIM * 4);
    float*    WcS    = (float*)   carve(cur, (size_t)G_DIM * N_CLS * 4);
    float*    WdS    = (float*)   carve(cur, (size_t)N_DOM * G_DIM * 2 * 4);
    float*    d2     = (float*)   carve(cur, (size_t)N * N * 4);   // 151 MB

    // ---- 1) conversions (+ weight transposes) -------------------------------
    {
        int n;
        n = N * D_IN;
        cvt_f32_to_f16<<<(n + 255) / 256, 256, 0, stream>>>(x, x_h, n);
        n = D_IN * H1_DIM;
        cvt_transpose_f16<<<(n + 255) / 256, 256, 0, stream>>>(fcW0, W0t_h, D_IN, H1_DIM);
        n = H1_DIM * H2_DIM;
        cvt_transpose_f16<<<(n + 255) / 256, 256, 0, stream>>>(fcW1, W1t_h, H1_DIM, H2_DIM);
        n = H2_DIM * G_DIM;
        cvt_transpose_f16<<<(n + 255) / 256, 256, 0, stream>>>(gatW, gWt_h, H2_DIM, G_DIM);
    }

    // ---- 2) h1 = relu(x @ W0 + b0) ------------------------------------------
    wmma_gemm32<<<dim3(N / 32, H1_DIM / 32), 32, 0, stream>>>(
        x_h, W0t_h, nullptr, h1_h, fcb0, N, H1_DIM, D_IN, /*relu=*/1);

    // ---- 3) h2 = relu(h1 @ W1 + b1) -----------------------------------------
    wmma_gemm32<<<dim3(N / 32, H2_DIM / 32), 32, 0, stream>>>(
        h1_h, W1t_h, h2_f, h2_h, fcb1, N, H2_DIM, H1_DIM, /*relu=*/1);

    // ---- 4) squared norms ----------------------------------------------------
    row_sqnorm<<<(N + 255) / 256, 256, 0, stream>>>(h2_f, sq, N, H2_DIM);

    // ---- 5) pairwise distances (dominant GEMM) ------------------------------
    wmma_dist<<<dim3(N / 32, N / 32), 32, 0, stream>>>(h2_h, sq, d2, N, H2_DIM);

    // ---- 6) per-row top-32 (async LDS staging) ------------------------------
    topk_rows<<<N, 256, 0, stream>>>(d2, knn, N, K_NN);

    // ---- 7) Wh = h2 @ gat_W --------------------------------------------------
    wmma_gemm32<<<dim3(N / 32, G_DIM / 32), 32, 0, stream>>>(
        h2_h, gWt_h, Wh_f, nullptr, nullptr, N, G_DIM, H2_DIM, /*relu=*/0);

    // ---- 8) attention projections -------------------------------------------
    attn_proj<<<(N + 255) / 256, 256, 0, stream>>>(Wh_f, gata, f1, f2, N, G_DIM);

    // ---- 9) sparse softmax + elu aggregate ----------------------------------
    gat_aggregate<<<N, 64, 0, stream>>>(Wh_f, f1, f2, knn, hp, N, G_DIM, K_NN);

    // ---- 10) heads -----------------------------------------------------------
    reduce_heads<<<4, 256, 0, stream>>>(clsW, dscW, WcS, WdS);
    head_kernel<<<N, 32, 0, stream>>>(hp, WcS, WdS, clsb, dscb, out, N, G_DIM);
}
